// LinearLoopLayer_21251498180719
// MI455X (gfx1250) — matrix operand order repaired and checked by reading, not compiled, not run on hardware
//
#include <hip/hip_runtime.h>
#include <hip/hip_bf16.h>

// MI455X (gfx1250) f32 linear layer: f16 WMMA + Tensor Data Mover pipeline.
//
// out[8192,2048] = x[8192,2048] @ W^T[2048,2048] + bias
//
//  - Pass 1: f32 -> f16 into workspace (40 MiB; f16 working set fits in L2).
//  - Pass 2: block tile 128x128, 4 waves, each wave owns 64x64 = 4x4 WMMA
//    tiles (1:1 ds_load_b128 : v_wmma ratio). A/B K-slabs (128x64 f16) are
//    staged to LDS by the TDM (tensor_load_to_lds, TENSORcnt, double
//    buffered) with pad-on-load (+16B per 128B row -> 144B row stride,
//    conflict-free b128 LDS fragment reads).

typedef __attribute__((ext_vector_type(16))) _Float16 v16h;
typedef __attribute__((ext_vector_type(8)))  _Float16 v8h;
typedef __attribute__((ext_vector_type(4)))  _Float16 v4h;
typedef __attribute__((ext_vector_type(8)))  float    v8f;
typedef __attribute__((ext_vector_type(4)))  unsigned int u32x4;
typedef __attribute__((ext_vector_type(8)))  int          i32x8;
typedef __attribute__((ext_vector_type(4)))  int          i32x4;

#define IN_F  2048
#define OUT_F 2048
#define BATCH 8192

#define KT        64              // K-slab staged per TDM transfer
#define TILE_ROWS 128             // rows per staged A/B slab (M or N extent)
#define ROW_H     72              // LDS row stride in halves (64 data + 4 DW pad)
#define SLAB_H    (TILE_ROWS * ROW_H)   // halves per slab buffer

// ---------------------------------------------------------------------------
// f32 -> f16 conversion, 4 elements per thread
// ---------------------------------------------------------------------------
__global__ __launch_bounds__(256) void cvt_f32_to_f16(
    const float* __restrict__ in, _Float16* __restrict__ out, int n4) {
  int i = blockIdx.x * blockDim.x + threadIdx.x;
  if (i < n4) {
    const float4 f = reinterpret_cast<const float4*>(in)[i];
    v4h h;
    h.x = (_Float16)f.x;
    h.y = (_Float16)f.y;
    h.z = (_Float16)f.z;
    h.w = (_Float16)f.w;
    reinterpret_cast<v4h*>(out)[i] = h;
  }
}

// ---------------------------------------------------------------------------
// TDM: stage one 128(rows) x 64(K) f16 tile, row-major, into LDS with
// 4-DWORD padding after every 32 DWORDs (row stride 144B in LDS).
// D# bitfields per CDNA5 ISA ch.8 (group0/group1; groups 2-3 unused -> 0).
// ---------------------------------------------------------------------------
__device__ __forceinline__ void tdm_load_tile(const _Float16* gsrc,
                                              unsigned lds_byte_off,
                                              unsigned tensor_rows) {
  const unsigned long long ga = (unsigned long long)(uintptr_t)gsrc;
  // Group 0: count=1 (valid user D#), lds_addr, global_addr[56:0], type=2.
  const u32x4 g0 = {
      1u,
      lds_byte_off,
      (unsigned)ga,
      ((unsigned)(ga >> 32) & 0x01FFFFFFu) | (2u << 30)
  };
  // Group 1:
  //  [17:16] data_size=1 (2B)  [20] pad_enable
  //  [24:22] pad_interval=4 (pad every 32 DWORDs = one 128B row)
  //  [31:25] pad_amount=3 (4 DWORDs = 16B)
  //  tensor_dim0 = IN_F (K extent), tensor_dim1 = tensor_rows
  //  tile_dim0 = 64 (K), tile_dim1 = 128 (rows), tensor_dim0_stride = IN_F
  const unsigned dim0 = (unsigned)IN_F;
  const unsigned dim1 = tensor_rows;
  i32x8 g1;
  g1[0] = (int)((1u << 16) | (1u << 20) | (4u << 22) | (3u << 25));
  g1[1] = (int)((dim0 & 0xFFFFu) << 16);
  g1[2] = (int)(((dim0 >> 16) & 0xFFFFu) | ((dim1 & 0xFFFFu) << 16));
  g1[3] = (int)(((dim1 >> 16) & 0xFFFFu) | ((unsigned)KT << 16));
  g1[4] = (int)TILE_ROWS;          // tile_dim1; tile_dim2 = 0
  g1[5] = (int)IN_F;               // tensor_dim0_stride[31:0]
  g1[6] = 0;                       // stride[47:32] = 0, dim1_stride lo = 0
  g1[7] = 0;
  const i32x4 gz4 = {0, 0, 0, 0};
  const i32x8 gz8 = {0, 0, 0, 0, 0, 0, 0, 0};
  // 6-arg variant (clang-23 / therock-10.0 headers).
  __builtin_amdgcn_tensor_load_to_lds(g0, g1, gz4, gz4, gz8, 0);
}

// ---------------------------------------------------------------------------
// WMMA GEMM with TDM->LDS double-buffered pipeline.
// Block: 128 threads = 4 waves (2x2), block tile 128(M) x 128(N).
// Wave tile: 64x64 = 4x4 v_wmma_f32_16x16x32_f16 tiles.
// ---------------------------------------------------------------------------
__global__ __launch_bounds__(128) void gemm_f16_wmma_tdm(
    const _Float16* __restrict__ A,   // [BATCH][IN_F]
    const _Float16* __restrict__ W,   // [OUT_F][IN_F]
    const float*    __restrict__ bias,
    float*          __restrict__ out) {
  extern __shared__ _Float16 smem[];
  _Float16* smA = smem;                 // 2 x SLAB_H halves
  _Float16* smB = smem + 2 * SLAB_H;    // 2 x SLAB_H halves

  const int lane = threadIdx.x & 31;
  const int wave = threadIdx.x >> 5;
  const int wm   = wave & 1;            // 2 waves along M
  const int wn   = wave >> 1;           // 2 waves along N
  const int l16  = lane & 15;
  const int kg   = lane >> 4;

  const int blockM = blockIdx.y * 128;
  const int blockN = blockIdx.x * 128;
  const bool isW0 = (wave == 0);

  v8f acc[4][4] = {};

  // Prologue: stage slab 0.
  if (isW0) {
    tdm_load_tile(A + (size_t)blockM * IN_F, (unsigned)(uintptr_t)smA, BATCH);
    tdm_load_tile(W + (size_t)blockN * IN_F, (unsigned)(uintptr_t)smB, OUT_F);
  }

  int buf = 0;
  for (int k0 = 0; k0 < IN_F; k0 += KT) {
    const bool hasNext = (k0 + KT) < IN_F;
    if (isW0) {
      if (hasNext) {
        const int nb = buf ^ 1;
        tdm_load_tile(A + (size_t)blockM * IN_F + (k0 + KT),
                      (unsigned)(uintptr_t)(smA + nb * SLAB_H), BATCH);
        tdm_load_tile(W + (size_t)blockN * IN_F + (k0 + KT),
                      (unsigned)(uintptr_t)(smB + nb * SLAB_H), OUT_F);
        // In-order TENSORcnt: <=2 outstanding means current slab's pair done.
        __builtin_amdgcn_s_wait_tensorcnt(2);
      } else {
        __builtin_amdgcn_s_wait_tensorcnt(0);
      }
    }
    __syncthreads();   // current slab visible to all waves

    const _Float16* sA = smA + buf * SLAB_H + (size_t)(wm * 64) * ROW_H;
    const _Float16* sB = smB + buf * SLAB_H + (size_t)(wn * 64) * ROW_H;

#pragma unroll
    for (int ksub = 0; ksub < KT; ksub += 32) {
      union { v16h v; v8h h[2]; } a[4], b[4];
      // A 16x32 f16 fragment: lane group kg holds K = 8*kg+0..7 (e0..7) and
      // 16+8*kg+0..7 (e8..15).
      const int aOff = ksub + 8 * kg;
#pragma unroll
      for (int i = 0; i < 4; ++i) {
        const _Float16* r = sA + (size_t)(i * 16 + l16) * ROW_H;
        a[i].h[0] = *reinterpret_cast<const v8h*>(r + aOff);
        a[i].h[1] = *reinterpret_cast<const v8h*>(r + aOff + 16);
      }
      // B 32x16 f16 fragment: lane group kg holds K = 8*kg+0..7 (e0..7) and
      // 16+8*kg+0..7 (e8..15).
      const int bOff = ksub + 8 * kg;
#pragma unroll
      for (int j = 0; j < 4; ++j) {
        const _Float16* r = sB + (size_t)(j * 16 + l16) * ROW_H;
        b[j].h[0] = *reinterpret_cast<const v8h*>(r + bOff);
        b[j].h[1] = *reinterpret_cast<const v8h*>(r + bOff + 16);
      }
#pragma unroll
      for (int i = 0; i < 4; ++i)
#pragma unroll
        for (int j = 0; j < 4; ++j)
          acc[i][j] = __builtin_amdgcn_wmma_f32_16x16x32_f16(
              false, a[i].v, false, b[j].v, (short)0, acc[i][j], false, false);
    }

    __syncthreads();   // all reads done before next iteration overwrites buf^1
    buf ^= 1;
  }

  // Epilogue: C/D layout — VGPR r holds M = r + 8*(lane>=16), N = lane&15.
  const int mBase = blockM + wm * 64;
  const int nBase = blockN + wn * 64;
#pragma unroll
  for (int j = 0; j < 4; ++j) {
    const int col = nBase + j * 16 + l16;
    const float bv = bias[col];
#pragma unroll
    for (int i = 0; i < 4; ++i) {
      const int rowBase = mBase + i * 16 + 8 * kg;
#pragma unroll
      for (int r = 0; r < 8; ++r) {
        out[(size_t)(rowBase + r) * OUT_F + col] = acc[i][j][r] + bv;
      }
    }
  }
}

// ---------------------------------------------------------------------------
// Fallback (only if workspace is unexpectedly small): naive f32 VALU GEMM.
// ---------------------------------------------------------------------------
__global__ __launch_bounds__(256) void gemm_naive_f32(
    const float* __restrict__ x, const float* __restrict__ w,
    const float* __restrict__ b, float* __restrict__ out) {
  size_t idx = (size_t)blockIdx.x * blockDim.x + threadIdx.x;
  if (idx >= (size_t)BATCH * OUT_F) return;
  const int row = (int)(idx / OUT_F);
  const int col = (int)(idx % OUT_F);
  const float* xr = x + (size_t)row * IN_F;
  const float* wr = w + (size_t)col * IN_F;
  float s = b[col];
  for (int k = 0; k < IN_F; ++k) s = fmaf(xr[k], wr[k], s);
  out[idx] = s;
}

extern "C" void kernel_launch(void* const* d_in, const int* in_sizes, int n_in,
                              void* d_out, int out_size, void* d_ws, size_t ws_size,
                              hipStream_t stream) {
  const float* x    = (const float*)d_in[0];
  const float* w    = (const float*)d_in[1];
  const float* bias = (const float*)d_in[2];
  float* out = (float*)d_out;

  const size_t nx = (size_t)BATCH * IN_F;   // 16,777,216
  const size_t nw = (size_t)OUT_F * IN_F;   //  4,194,304
  const size_t need = (nx + nw) * sizeof(_Float16);  // 40 MiB

  if (ws_size >= need) {
    _Float16* Ah = (_Float16*)d_ws;
    _Float16* Wh = Ah + nx;

    const int nx4 = (int)(nx / 4);
    const int nw4 = (int)(nw / 4);
    cvt_f32_to_f16<<<(nx4 + 255) / 256, 256, 0, stream>>>(x, Ah, nx4);
    cvt_f32_to_f16<<<(nw4 + 255) / 256, 256, 0, stream>>>(w, Wh, nw4);

    const size_t ldsBytes = (size_t)4 * SLAB_H * sizeof(_Float16);  // 73728
    dim3 grid(OUT_F / 128, BATCH / 128);  // (16, 64)
    gemm_f16_wmma_tdm<<<grid, 128, ldsBytes, stream>>>(Ah, Wh, bias, out);
  } else {
    const size_t total = (size_t)BATCH * OUT_F;
    gemm_naive_f32<<<(unsigned)((total + 255) / 256), 256, 0, stream>>>(x, w, bias, out);
  }
}
